// PhoBERT_CRF_KGAN_25065429139540
// MI455X (gfx1250) — matrix operand order repaired and checked by reading, not compiled, not run on hardware
//
#include <hip/hip_runtime.h>
#include <hip/hip_bf16.h>

// ---------------------------------------------------------------------------
// PhoBERT-CRF-KGAN forward pass on gfx1250 (CDNA5, wave32, WMMA).
// Dense GEMMs: v_wmma_f32_16x16x32_f16 (f16 in, f32 accum), LDS-tiled,
// double-buffered with async global->LDS copies when the toolchain has them.
// ---------------------------------------------------------------------------

typedef _Float16 f16_t;
typedef __attribute__((ext_vector_type(16))) _Float16 v16h;
typedef __attribute__((ext_vector_type(8)))  _Float16 v8h;
typedef __attribute__((ext_vector_type(8)))  float    v8f;
typedef __attribute__((ext_vector_type(4)))  int      v4i_t;

#define NB   32      // batch
#define SL   256     // sequence length
#define HD   768     // hidden (bert part)
#define KGD  200     // kg part
#define DD   968     // HD + KGD
#define FFD  3072
#define LHD  128     // lstm hidden per direction
#define NT   11      // tags
#define NR   8192    // NB * SL rows

#define KG_P 224     // KGD padded to 32
#define D_P  992     // DD padded to 32

#if defined(__HIP_DEVICE_COMPILE__) &&                                   \
    __has_builtin(__builtin_amdgcn_global_load_async_to_lds_b128) &&      \
    __has_builtin(__builtin_amdgcn_s_wait_asynccnt)
#define USE_ASYNC 1
#else
#define USE_ASYNC 0
#endif

#define GLOBAL_AS __attribute__((address_space(1)))
#define LDS_AS    __attribute__((address_space(3)))

__device__ __forceinline__ float sigf(float x) { return 1.0f / (1.0f + expf(-x)); }

__device__ __forceinline__ void async_copy_b128(const f16_t* g, f16_t* l) {
#if USE_ASYNC
    __builtin_amdgcn_global_load_async_to_lds_b128(
        (GLOBAL_AS v4i_t*)g, (LDS_AS v4i_t*)l, 0, 0);
#else
    (void)g; (void)l;
#endif
}

// ---------------------------------------------------------------------------
// pack: f32 (srcRows x srcCols, row stride srcStride) -> f16 (dstRows x dstCols)
// zero-fills K padding and row padding.
// ---------------------------------------------------------------------------
__global__ void pack_f16_kernel(const float* __restrict__ src, long srcStride,
                                int srcRows, int srcCols,
                                f16_t* __restrict__ dst, int dstCols, long total) {
    long idx = (long)blockIdx.x * blockDim.x + threadIdx.x;
    if (idx >= total) return;
    int r = (int)(idx / dstCols);
    int c = (int)(idx % dstCols);
    float v = (r < srcRows && c < srcCols) ? src[(size_t)r * srcStride + c] : 0.0f;
    dst[idx] = (f16_t)v;
}

// ---------------------------------------------------------------------------
// Generic WMMA GEMM: C(MxN) = A(MxKp) * B(NpxKp)^T (+bias[N]) (+resid) (relu)
// A, B f16 row-major, K-contiguous, Kp multiple of 32.
// Block = 256 threads (8 waves), macro tile 128(M) x 64(N), wave tile 32x32.
// Double-buffered LDS; tile k+1 streams via GLOBAL_LOAD_ASYNC_TO_LDS while
// tile k is consumed by WMMA (fallback: synchronous staging).
// ---------------------------------------------------------------------------
template <bool RELU, bool RESID, bool F32OUT, bool F16OUT>
__global__ __launch_bounds__(256)
void gemm_wmma_kernel(const f16_t* __restrict__ A, const f16_t* __restrict__ B,
                      const float* __restrict__ bias, const float* __restrict__ resid,
                      float* __restrict__ Cf32, f16_t* __restrict__ Cf16,
                      int M, int N, int Kp, int ldc, int ldc16) {
    __shared__ __align__(16) f16_t As[2][128 * 32];
    __shared__ __align__(16) f16_t Bs[2][64 * 32];

    const int tid  = threadIdx.x;
    const int wave = tid >> 5;
    const int lane = tid & 31;
    const int laneM    = lane & 15;
    const int laneHalf = lane >> 4;

    const int bm0 = blockIdx.y * 128;
    const int bn0 = blockIdx.x * 64;
    const int wm0 = (wave >> 1) * 32;  // 0,32,64,96
    const int wn0 = (wave & 1) * 32;   // 0,32

    v8f acc[2][2] = {};

    // per-thread staging coordinates (A: 2 x b128, B: 1 x b128 per thread)
    const int ra0 = tid >> 2;
    const int ra1 = (tid + 256) >> 2;
    const int ca  = (tid & 3) * 8;

    auto compute_tile = [&](const f16_t* Asb, const f16_t* Bsb) {
        v16h afrag[2], bfrag[2];
        #pragma unroll
        for (int mt = 0; mt < 2; ++mt) {
            int row = wm0 + mt * 16 + laneM;
            v8h* ap = (v8h*)&afrag[mt];
            ap[0] = *(const v8h*)&Asb[row * 32 + laneHalf * 8];
            ap[1] = *(const v8h*)&Asb[row * 32 + laneHalf * 8 + 16];
        }
        #pragma unroll
        for (int nt = 0; nt < 2; ++nt) {
            int row = wn0 + nt * 16 + laneM;
            v8h* bp = (v8h*)&bfrag[nt];
            bp[0] = *(const v8h*)&Bsb[row * 32 + laneHalf * 8];
            bp[1] = *(const v8h*)&Bsb[row * 32 + laneHalf * 8 + 16];
        }
        #pragma unroll
        for (int mt = 0; mt < 2; ++mt)
            #pragma unroll
            for (int nt = 0; nt < 2; ++nt)
                acc[mt][nt] = __builtin_amdgcn_wmma_f32_16x16x32_f16(
                    false, afrag[mt], false, bfrag[nt],
                    (short)0, acc[mt][nt], false, false);
    };

    const int nk = Kp / 32;

#if USE_ASYNC
    auto issue_tile = [&](int ki, int buf) {
        const int k0 = ki * 32;
        async_copy_b128(&A[(size_t)(bm0 + ra0) * Kp + k0 + ca], &As[buf][ra0 * 32 + ca]);
        async_copy_b128(&A[(size_t)(bm0 + ra1) * Kp + k0 + ca], &As[buf][ra1 * 32 + ca]);
        async_copy_b128(&B[(size_t)(bn0 + ra0) * Kp + k0 + ca], &Bs[buf][ra0 * 32 + ca]);
    };
    issue_tile(0, 0);
    for (int ki = 0; ki < nk; ++ki) {
        const int buf = ki & 1;
        if (ki + 1 < nk) {
            issue_tile(ki + 1, buf ^ 1);
            __builtin_amdgcn_s_wait_asynccnt(3);   // tile ki resident, ki+1 in flight
        } else {
            __builtin_amdgcn_s_wait_asynccnt(0);
        }
        __syncthreads();                            // all waves' tile-ki loads visible
        compute_tile(&As[buf][0], &Bs[buf][0]);
        __syncthreads();                            // compute done before buf reuse
    }
#else
    for (int ki = 0; ki < nk; ++ki) {
        const int k0 = ki * 32;
        *(v8h*)&As[0][ra0 * 32 + ca] = *(const v8h*)&A[(size_t)(bm0 + ra0) * Kp + k0 + ca];
        *(v8h*)&As[0][ra1 * 32 + ca] = *(const v8h*)&A[(size_t)(bm0 + ra1) * Kp + k0 + ca];
        *(v8h*)&Bs[0][ra0 * 32 + ca] = *(const v8h*)&B[(size_t)(bn0 + ra0) * Kp + k0 + ca];
        __syncthreads();
        compute_tile(&As[0][0], &Bs[0][0]);
        __syncthreads();
    }
#endif

    // epilogue
    #pragma unroll
    for (int mt = 0; mt < 2; ++mt) {
        #pragma unroll
        for (int nt = 0; nt < 2; ++nt) {
            int n = bn0 + wn0 + nt * 16 + laneM;
            if (n >= N) continue;
            float bv = bias[n];
            #pragma unroll
            for (int r = 0; r < 8; ++r) {
                int m = bm0 + wm0 + mt * 16 + r + laneHalf * 8;
                float v = acc[mt][nt][r] + bv;
                if (RESID) v += resid[(size_t)m * ldc + n];
                if (RELU)  v = v > 0.0f ? v : 0.0f;
                if (F32OUT) Cf32[(size_t)m * ldc + n] = v;
                if (F16OUT) Cf16[(size_t)m * ldc16 + n] = (f16_t)v;
            }
        }
    }
}

// ---------------------------------------------------------------------------
// LayerNorm (+affine) + tanh, per row of HD columns. Block per row.
// ---------------------------------------------------------------------------
__global__ __launch_bounds__(256)
void ln_tanh_kernel(float* __restrict__ h, const float* __restrict__ g,
                    const float* __restrict__ b, int cols) {
    __shared__ float red[256];
    float* hr = h + (size_t)blockIdx.x * cols;
    int tid = threadIdx.x;

    float s = 0.0f;
    for (int c = tid; c < cols; c += 256) s += hr[c];
    red[tid] = s; __syncthreads();
    for (int st = 128; st > 0; st >>= 1) {
        if (tid < st) red[tid] += red[tid + st];
        __syncthreads();
    }
    float mu = red[0] / cols;
    __syncthreads();

    float vs = 0.0f;
    for (int c = tid; c < cols; c += 256) { float d = hr[c] - mu; vs += d * d; }
    red[tid] = vs; __syncthreads();
    for (int st = 128; st > 0; st >>= 1) {
        if (tid < st) red[tid] += red[tid + st];
        __syncthreads();
    }
    float inv = rsqrtf(red[0] / cols + 1e-5f);
    __syncthreads();

    for (int c = tid; c < cols; c += 256)
        hr[c] = tanhf((hr[c] - mu) * inv * g[c] + b[c]);
}

// ---------------------------------------------------------------------------
// attention score: scores[row] = dot(h[row,:], w) + b2[0]. Block per row.
// ---------------------------------------------------------------------------
__global__ __launch_bounds__(256)
void attn_score_kernel(const float* __restrict__ h, const float* __restrict__ w,
                       const float* __restrict__ b2, float* __restrict__ scores,
                       int cols) {
    __shared__ float red[256];
    int tid = threadIdx.x;
    const float* hr = h + (size_t)blockIdx.x * cols;
    float s = 0.0f;
    for (int c = tid; c < cols; c += 256) s += hr[c] * w[c];
    red[tid] = s; __syncthreads();
    for (int st = 128; st > 0; st >>= 1) {
        if (tid < st) red[tid] += red[tid + st];
        __syncthreads();
    }
    if (tid == 0) scores[blockIdx.x] = red[0] + b2[0];
}

// ---------------------------------------------------------------------------
// softmax over the sequence dim per batch (S == 256 == blockDim.x).
// ---------------------------------------------------------------------------
__global__ __launch_bounds__(256)
void softmax_seq_kernel(const float* __restrict__ scores, float* __restrict__ attw) {
    __shared__ float red[256];
    int b = blockIdx.x, tid = threadIdx.x;
    float v = scores[(size_t)b * SL + tid];
    red[tid] = v; __syncthreads();
    for (int st = 128; st > 0; st >>= 1) {
        if (tid < st) red[tid] = fmaxf(red[tid], red[tid + st]);
        __syncthreads();
    }
    float mx = red[0]; __syncthreads();
    float e = expf(v - mx);
    red[tid] = e; __syncthreads();
    for (int st = 128; st > 0; st >>= 1) {
        if (tid < st) red[tid] += red[tid + st];
        __syncthreads();
    }
    attw[(size_t)b * SL + tid] = e / red[0];
}

// ---------------------------------------------------------------------------
// combined = [bert, kg*(attw+1)]; also emits f16 copy with K padding D_P.
// ---------------------------------------------------------------------------
__global__ void build_combined_kernel(const float* __restrict__ emb,
                                      const float* __restrict__ attw,
                                      float* __restrict__ comb,
                                      f16_t* __restrict__ comb16, long total) {
    long idx = (long)blockIdx.x * blockDim.x + threadIdx.x;
    if (idx >= total) return;
    int row = (int)(idx / D_P);
    int c   = (int)(idx % D_P);
    float v = 0.0f;
    if (c < DD) {
        float e = emb[(size_t)row * DD + c];
        v = (c < HD) ? e : e * (attw[row] + 1.0f);
        comb[(size_t)row * DD + c] = v;
    }
    comb16[idx] = (f16_t)v;
}

// ---------------------------------------------------------------------------
// BiLSTM recurrence (one direction per block; grid.x == 2).
// Per step: G(32x512) = Hprev(32x128) @ Whh^T(128x512) via WMMA (8 waves),
// + pre, then gate nonlinearity; h/c state lives in LDS.
// ---------------------------------------------------------------------------
__global__ __launch_bounds__(256)
void lstm_recurrent_kernel(const float* __restrict__ preF, const float* __restrict__ preB,
                           const f16_t* __restrict__ WhhF, const f16_t* __restrict__ WhhB,
                           float* __restrict__ Gbuf, float* __restrict__ out) {
    const int dir = blockIdx.x;
    const float* pre  = dir ? preB : preF;
    const f16_t* Whh  = dir ? WhhB : WhhF;
    float*       G    = Gbuf + (size_t)dir * NB * 512;

    __shared__ __align__(16) f16_t Hs[NB * LHD];   // 8 KB
    __shared__ float Cs[NB * LHD];                 // 16 KB

    const int tid  = threadIdx.x;
    const int wave = tid >> 5;
    const int lane = tid & 31;
    const int laneM    = lane & 15;
    const int laneHalf = lane >> 4;

    for (int p = tid; p < NB * LHD; p += 256) { Hs[p] = (f16_t)0.0f; Cs[p] = 0.0f; }
    __syncthreads();

    for (int s = 0; s < SL; ++s) {
        const int t = dir ? (SL - 1 - s) : s;

        // ---- G = Hprev @ Whh^T : wave owns gate columns [wave*64, wave*64+64)
        #pragma unroll
        for (int mt = 0; mt < 2; ++mt) {
            const int arow = mt * 16 + laneM;           // batch index 0..31
            #pragma unroll
            for (int nt = 0; nt < 4; ++nt) {
                const int n0 = wave * 64 + nt * 16;
                v8f acc = {};
                #pragma unroll
                for (int kk = 0; kk < 4; ++kk) {
                    const int k0 = kk * 32;
                    v16h a, bfrag;
                    v8h* ap = (v8h*)&a;
                    ap[0] = *(const v8h*)&Hs[arow * LHD + k0 + laneHalf * 8];
                    ap[1] = *(const v8h*)&Hs[arow * LHD + k0 + laneHalf * 8 + 16];
                    v8h* bp = (v8h*)&bfrag;
                    const f16_t* brow = &Whh[(size_t)(n0 + laneM) * LHD + k0 + laneHalf * 8];
                    bp[0] = *(const v8h*)brow;
                    bp[1] = *(const v8h*)(brow + 16);
                    acc = __builtin_amdgcn_wmma_f32_16x16x32_f16(
                        false, a, false, bfrag, (short)0, acc, false, false);
                }
                #pragma unroll
                for (int r = 0; r < 8; ++r) {
                    const int m = mt * 16 + r + laneHalf * 8;  // batch
                    const int g = n0 + laneM;                  // gate col
                    G[m * 512 + g] = acc[r] + pre[((size_t)m * SL + t) * 512 + g];
                }
            }
        }
        __threadfence_block();
        __syncthreads();

        // ---- gates: 32*128 cells, 16 per thread
        for (int p = tid; p < NB * LHD; p += 256) {
            const int b = p >> 7, j = p & 127;
            const float gi = G[b * 512 + j];
            const float gf = G[b * 512 + 128 + j];
            const float gg = G[b * 512 + 256 + j];
            const float go = G[b * 512 + 384 + j];
            const float c  = sigf(gf) * Cs[p] + sigf(gi) * tanhf(gg);
            const float h  = sigf(go) * tanhf(c);
            Cs[p] = c;
            Hs[p] = (f16_t)h;
            out[((size_t)b * SL + t) * (2 * LHD) + dir * LHD + j] = h;
        }
        __threadfence_block();
        __syncthreads();
    }
}

// ---------------------------------------------------------------------------
// emissions = x(NR x 256) @ cls_W^T(11 x 256) + cls_b
// ---------------------------------------------------------------------------
__global__ void emissions_kernel(const float* __restrict__ x, const float* __restrict__ W,
                                 const float* __restrict__ b, float* __restrict__ em) {
    int idx = blockIdx.x * blockDim.x + threadIdx.x;
    if (idx >= NR * NT) return;
    int r = idx / NT, tg = idx % NT;
    const float* xr = x + (size_t)r * (2 * LHD);
    const float* wr = W + (size_t)tg * (2 * LHD);
    float s = b[tg];
    for (int k = 0; k < 2 * LHD; ++k) s += xr[k] * wr[k];
    em[idx] = s;
}

// ---------------------------------------------------------------------------
// CRF negative log-likelihood, mean over batch. One block, 352 = 32*11 threads.
// ---------------------------------------------------------------------------
__global__ __launch_bounds__(352)
void crf_nll_kernel(const float* __restrict__ em, const int* __restrict__ labels,
                    const int* __restrict__ mask, const float* __restrict__ trans,
                    float* __restrict__ out) {
    __shared__ float tr[NT * NT];
    __shared__ float alpha[NB * NT];
    __shared__ float nalpha[NB * NT];
    __shared__ float gold[NB];
    __shared__ float nll[NB];
    const int tid = threadIdx.x;
    const int START = 9, END = 10;

    if (tid < NT * NT) tr[tid] = trans[tid];
    __syncthreads();

    const int b = tid / NT, j = tid % NT;
    if (tid < NB * NT)
        alpha[tid] = tr[START * NT + j] + em[((size_t)b * SL) * NT + j];

    if (tid < NB) {
        const int* lb = labels + (size_t)tid * SL;
        const int* mk = mask + (size_t)tid * SL;
        float emit = 0.0f, pair = 0.0f;
        int len = 0;
        for (int s = 0; s < SL; ++s) {
            float mf = (float)mk[s];
            len += mk[s];
            emit += em[((size_t)tid * SL + s) * NT + lb[s]] * mf;
            if (s > 0) pair += tr[lb[s - 1] * NT + lb[s]] * mf;
        }
        int last = lb[len - 1];
        gold[tid] = tr[START * NT + lb[0]] + emit + pair + tr[last * NT + END];
    }
    __syncthreads();

    for (int s = 1; s < SL; ++s) {
        if (tid < NB * NT) {
            float mx = -1e30f;
            for (int i = 0; i < NT; ++i)
                mx = fmaxf(mx, alpha[b * NT + i] + tr[i * NT + j]);
            float sum = 0.0f;
            for (int i = 0; i < NT; ++i)
                sum += expf(alpha[b * NT + i] + tr[i * NT + j] - mx);
            float nv = mx + logf(sum) + em[((size_t)b * SL + s) * NT + j];
            nalpha[tid] = mask[(size_t)b * SL + s] ? nv : alpha[tid];
        }
        __syncthreads();
        if (tid < NB * NT) alpha[tid] = nalpha[tid];
        __syncthreads();
    }

    if (tid < NB) {
        float mx = -1e30f;
        for (int i = 0; i < NT; ++i)
            mx = fmaxf(mx, alpha[tid * NT + i] + tr[i * NT + END]);
        float sum = 0.0f;
        for (int i = 0; i < NT; ++i)
            sum += expf(alpha[tid * NT + i] + tr[i * NT + END] - mx);
        nll[tid] = mx + logf(sum) - gold[tid];
    }
    __syncthreads();
    if (tid == 0) {
        float s = 0.0f;
        for (int i = 0; i < NB; ++i) s += nll[i];
        out[0] = s / NB;
    }
}

// ---------------------------------------------------------------------------
// Host orchestration
// ---------------------------------------------------------------------------
static inline void launch_pack(const float* src, long stride, int srows, int scols,
                               f16_t* dst, int dcols, int drows, hipStream_t st) {
    long total = (long)drows * dcols;
    int grid = (int)((total + 255) / 256);
    pack_f16_kernel<<<grid, 256, 0, st>>>(src, stride, srows, scols, dst, dcols, total);
}

extern "C" void kernel_launch(void* const* d_in, const int* in_sizes, int n_in,
                              void* d_out, int out_size, void* d_ws, size_t ws_size,
                              hipStream_t stream) {
    const float* emb    = (const float*)d_in[0];
    const int*   amask  = (const int*)d_in[1];
    const int*   labels = (const int*)d_in[2];
    const float* kg_W1  = (const float*)d_in[3];
    const float* kg_b1  = (const float*)d_in[4];
    const float* ln_g   = (const float*)d_in[5];
    const float* ln_b   = (const float*)d_in[6];
    const float* kg_W2  = (const float*)d_in[7];
    const float* kg_b2  = (const float*)d_in[8];
    const float* ff_W1  = (const float*)d_in[9];
    const float* ff_b1  = (const float*)d_in[10];
    const float* ff_W2  = (const float*)d_in[11];
    const float* ff_b2  = (const float*)d_in[12];
    const float* Wih0f  = (const float*)d_in[13];
    const float* Whh0f  = (const float*)d_in[14];
    const float* b0f    = (const float*)d_in[15];
    const float* Wih0b  = (const float*)d_in[16];
    const float* Whh0b  = (const float*)d_in[17];
    const float* b0b    = (const float*)d_in[18];
    const float* Wih1f  = (const float*)d_in[19];
    const float* Whh1f  = (const float*)d_in[20];
    const float* b1f    = (const float*)d_in[21];
    const float* Wih1b  = (const float*)d_in[22];
    const float* Whh1b  = (const float*)d_in[23];
    const float* b1b    = (const float*)d_in[24];
    const float* cls_W  = (const float*)d_in[25];
    const float* cls_b  = (const float*)d_in[26];
    const float* transi = (const float*)d_in[27];

    // ---- workspace bump allocator (256B aligned) ----
    char* ws = (char*)d_ws;
    size_t off = 0;
    auto alloc = [&](size_t bytes) -> void* {
        void* p = ws + off;
        off += (bytes + 255) & ~(size_t)255;
        return p;
    };
    f16_t* kg16      = (f16_t*)alloc((size_t)NR * KG_P * 2);
    f16_t* kgW1_16   = (f16_t*)alloc((size_t)HD * KG_P * 2);
    float* hbuf      = (float*)alloc((size_t)NR * HD * 4);
    float* scores    = (float*)alloc((size_t)NR * 4);
    float* attw      = (float*)alloc((size_t)NR * 4);
    float* comb      = (float*)alloc((size_t)NR * DD * 4);
    f16_t* comb16    = (f16_t*)alloc((size_t)NR * D_P * 2);
    f16_t* ffW1_16   = (f16_t*)alloc((size_t)FFD * D_P * 2);
    f16_t* act1_16   = (f16_t*)alloc((size_t)NR * FFD * 2);
    f16_t* ffW2_16   = (f16_t*)alloc((size_t)1024 * FFD * 2);   // N padded to 1024
    float* xbuf      = (float*)alloc((size_t)NR * DD * 4);
    f16_t* x16       = (f16_t*)alloc((size_t)NR * D_P * 2);
    f16_t* Wih0f_16  = (f16_t*)alloc((size_t)512 * D_P * 2);
    f16_t* Wih0b_16  = (f16_t*)alloc((size_t)512 * D_P * 2);
    f16_t* Whh0f_16  = (f16_t*)alloc((size_t)512 * LHD * 2);
    f16_t* Whh0b_16  = (f16_t*)alloc((size_t)512 * LHD * 2);
    f16_t* Whh1f_16  = (f16_t*)alloc((size_t)512 * LHD * 2);
    f16_t* Whh1b_16  = (f16_t*)alloc((size_t)512 * LHD * 2);
    float* pre0f     = (float*)alloc((size_t)NR * 512 * 4);
    float* pre0b     = (float*)alloc((size_t)NR * 512 * 4);
    float* Gbuf      = (float*)alloc((size_t)2 * NB * 512 * 4);
    float* out0      = (float*)alloc((size_t)NR * 256 * 4);
    f16_t* out0_16   = (f16_t*)alloc((size_t)NR * 256 * 2);
    f16_t* Wih1f_16  = (f16_t*)alloc((size_t)512 * 256 * 2);
    f16_t* Wih1b_16  = (f16_t*)alloc((size_t)512 * 256 * 2);
    float* pre1f     = (float*)alloc((size_t)NR * 512 * 4);
    float* pre1b     = (float*)alloc((size_t)NR * 512 * 4);
    float* out1      = (float*)alloc((size_t)NR * 256 * 4);
    float* embuf     = (float*)alloc((size_t)NR * NT * 4);

    // ---- 1) KG projection: h = kg @ kg_W1^T + kg_b1 ----
    launch_pack(emb + HD, DD, NR, KGD, kg16, KG_P, NR, stream);
    launch_pack(kg_W1, KGD, HD, KGD, kgW1_16, KG_P, HD, stream);
    gemm_wmma_kernel<false, false, true, false>
        <<<dim3(HD / 64, NR / 128), 256, 0, stream>>>(
            kg16, kgW1_16, kg_b1, nullptr, hbuf, nullptr, NR, HD, KG_P, HD, 0);

    // ---- 2) LN + tanh; attention over sequence; combined ----
    ln_tanh_kernel<<<NR, 256, 0, stream>>>(hbuf, ln_g, ln_b, HD);
    attn_score_kernel<<<NR, 256, 0, stream>>>(hbuf, kg_W2, kg_b2, scores, HD);
    softmax_seq_kernel<<<NB, 256, 0, stream>>>(scores, attw);
    {
        long total = (long)NR * D_P;
        build_combined_kernel<<<(int)((total + 255) / 256), 256, 0, stream>>>(
            emb, attw, comb, comb16, total);
    }

    // ---- 3) FFN: relu(comb @ W1^T + b1) @ W2^T + b2 + comb ----
    launch_pack(ff_W1, DD, FFD, DD, ffW1_16, D_P, FFD, stream);
    gemm_wmma_kernel<true, false, false, true>
        <<<dim3(FFD / 64, NR / 128), 256, 0, stream>>>(
            comb16, ffW1_16, ff_b1, nullptr, nullptr, act1_16, NR, FFD, D_P, 0, FFD);
    launch_pack(ff_W2, FFD, DD, FFD, ffW2_16, FFD, 1024, stream);
    gemm_wmma_kernel<false, true, true, false>
        <<<dim3(1024 / 64, NR / 128), 256, 0, stream>>>(
            act1_16, ffW2_16, ff_b2, comb, xbuf, nullptr, NR, DD, FFD, DD, 0);
    launch_pack(xbuf, DD, NR, DD, x16, D_P, NR, stream);

    // ---- 4) BiLSTM layer 0 ----
    launch_pack(Wih0f, DD, 512, DD, Wih0f_16, D_P, 512, stream);
    launch_pack(Wih0b, DD, 512, DD, Wih0b_16, D_P, 512, stream);
    launch_pack(Whh0f, LHD, 512, LHD, Whh0f_16, LHD, 512, stream);
    launch_pack(Whh0b, LHD, 512, LHD, Whh0b_16, LHD, 512, stream);
    gemm_wmma_kernel<false, false, true, false>
        <<<dim3(512 / 64, NR / 128), 256, 0, stream>>>(
            x16, Wih0f_16, b0f, nullptr, pre0f, nullptr, NR, 512, D_P, 512, 0);
    gemm_wmma_kernel<false, false, true, false>
        <<<dim3(512 / 64, NR / 128), 256, 0, stream>>>(
            x16, Wih0b_16, b0b, nullptr, pre0b, nullptr, NR, 512, D_P, 512, 0);
    lstm_recurrent_kernel<<<2, 256, 0, stream>>>(pre0f, pre0b, Whh0f_16, Whh0b_16,
                                                 Gbuf, out0);

    // ---- 5) BiLSTM layer 1 ----
    launch_pack(out0, 256, NR, 256, out0_16, 256, NR, stream);
    launch_pack(Wih1f, 256, 512, 256, Wih1f_16, 256, 512, stream);
    launch_pack(Wih1b, 256, 512, 256, Wih1b_16, 256, 512, stream);
    launch_pack(Whh1f, LHD, 512, LHD, Whh1f_16, LHD, 512, stream);
    launch_pack(Whh1b, LHD, 512, LHD, Whh1b_16, LHD, 512, stream);
    gemm_wmma_kernel<false, false, true, false>
        <<<dim3(512 / 64, NR / 128), 256, 0, stream>>>(
            out0_16, Wih1f_16, b1f, nullptr, pre1f, nullptr, NR, 512, 256, 512, 0);
    gemm_wmma_kernel<false, false, true, false>
        <<<dim3(512 / 64, NR / 128), 256, 0, stream>>>(
            out0_16, Wih1b_16, b1b, nullptr, pre1b, nullptr, NR, 512, 256, 512, 0);
    lstm_recurrent_kernel<<<2, 256, 0, stream>>>(pre1f, pre1b, Whh1f_16, Whh1b_16,
                                                 Gbuf, out1);

    // ---- 6) emissions + CRF NLL ----
    emissions_kernel<<<(NR * NT + 255) / 256, 256, 0, stream>>>(out1, cls_W, cls_b, embuf);
    crf_nll_kernel<<<1, 352, 0, stream>>>(embuf, labels, amask, transi, (float*)d_out);

    (void)in_sizes; (void)n_in; (void)out_size; (void)ws_size;
}
